// ModularFusedMoEKernel_81028853006988
// MI455X (gfx1250) — matrix operand order repaired and checked by reading, not compiled, not run on hardware
//
#include <hip/hip_runtime.h>
#include <hip/hip_bf16.h>
#include <math.h>

#define MDIM 2048
#define KDIM 2048
#define ENUM 8
#define IDIM 2816
#define TOPKN 2
#define NDIM (2*IDIM)
#define SLOTS (MDIM*TOPKN)
#define LROW 40   // LDS row stride in bf16 (80B) -> conflict-free b128 frag loads

typedef __bf16 bf16;
typedef __attribute__((ext_vector_type(8)))  __bf16 v8bf;
typedef __attribute__((ext_vector_type(16))) __bf16 v16bf;
typedef __attribute__((ext_vector_type(8)))  float  v8f;

union V16 { v16bf v; v8bf h[2]; };

__device__ __forceinline__ v8bf cvt8(float4 a, float4 b) {
    v8bf r;
    r[0] = (__bf16)a.x; r[1] = (__bf16)a.y; r[2] = (__bf16)a.z; r[3] = (__bf16)a.w;
    r[4] = (__bf16)b.x; r[5] = (__bf16)b.y; r[6] = (__bf16)b.z; r[7] = (__bf16)b.w;
    return r;
}

// ---------------- routing: bin (token,slot) pairs per expert ----------------
__global__ void moe_route(const int* __restrict__ topk_ids,
                          int* __restrict__ counts, int* __restrict__ lists) {
    int t = blockIdx.x * blockDim.x + threadIdx.x;
    if (t < SLOTS) {
        int e = topk_ids[t];
        int pos = atomicAdd(&counts[e], 1);
        lists[e * SLOTS + pos] = t;
    }
}

// ---------------- GEMM1: h[slot, 0:I] = silu(x@Wg^T) * (x@Wu^T) -------------
__global__ __launch_bounds__(128)
void moe_gemm1(const float* __restrict__ x, const float* __restrict__ w1,
               const int* __restrict__ counts, const int* __restrict__ lists,
               float* __restrict__ hbuf) {
    const int e = blockIdx.z;
    const int cnt = counts[e];
    const int row0 = blockIdx.y * 32;
    if (row0 >= cnt) return;
    const int n0 = blockIdx.x * 64;   // columns of I handled by this block

    __shared__ __align__(16) bf16 lA [32 * LROW];
    __shared__ __align__(16) bf16 lBg[64 * LROW];
    __shared__ __align__(16) bf16 lBu[64 * LROW];
    __shared__ int s_slot[32];
    __shared__ int s_tok[32];

    const int tid = threadIdx.x;
    if (tid < 32) {
        int r = row0 + tid;
        int slot = (r < cnt) ? lists[e * SLOTS + r] : -1;
        s_slot[tid] = slot;
        s_tok[tid]  = (slot >= 0) ? (slot / TOPKN) : 0;
    }
    __syncthreads();

    // loader mapping (128 threads)
    const int ar = tid >> 2;            // 0..31  A row
    const int ac = (tid & 3) * 8;       // 0,8,16,24
    const int br = tid >> 1;            // 0..63  B row (N dim)
    const int bc = (tid & 1) * 16;      // 0,16
    const size_t aBase = (size_t)s_tok[ar] * KDIM + ac;
    const size_t gBase = (size_t)e * NDIM * KDIM + (size_t)(n0 + br) * KDIM + bc;
    const size_t uBase = gBase + (size_t)IDIM * KDIM;

    float4 ra0, ra1, rg0, rg1, rg2, rg3, ru0, ru1, ru2, ru3;
    auto loadRegs = [&](int k0) {
        const float* pa = x + aBase + k0;
        ra0 = *(const float4*)(pa);     ra1 = *(const float4*)(pa + 4);
        const float* pg = w1 + gBase + k0;
        rg0 = *(const float4*)(pg);     rg1 = *(const float4*)(pg + 4);
        rg2 = *(const float4*)(pg + 8); rg3 = *(const float4*)(pg + 12);
        const float* pu = w1 + uBase + k0;
        ru0 = *(const float4*)(pu);     ru1 = *(const float4*)(pu + 4);
        ru2 = *(const float4*)(pu + 8); ru3 = *(const float4*)(pu + 12);
    };
    auto storeLds = [&]() {
        *(v8bf*)&lA [ar * LROW + ac]      = cvt8(ra0, ra1);
        *(v8bf*)&lBg[br * LROW + bc]      = cvt8(rg0, rg1);
        *(v8bf*)&lBg[br * LROW + bc + 8]  = cvt8(rg2, rg3);
        *(v8bf*)&lBu[br * LROW + bc]      = cvt8(ru0, ru1);
        *(v8bf*)&lBu[br * LROW + bc + 8]  = cvt8(ru2, ru3);
    };

    // wave tiling: 4 waves in 2x2 -> each wave 16 rows x 32 cols (gate+up)
    const int lane = tid & 31;
    const int wid  = tid >> 5;
    const int wr = wid >> 1, wc = wid & 1;
    const int l  = lane & 15, hi = lane >> 4;

    v8f accg0 = {}, accg1 = {}, accu0 = {}, accu1 = {};

    auto compute = [&]() {
        V16 af;
        const bf16* pA = &lA[(16 * wr + l) * LROW];
        af.h[0] = *(const v8bf*)(pA + hi * 16);       // K 0..7   / 16..23
        af.h[1] = *(const v8bf*)(pA + hi * 16 + 8);   // K 8..15  / 24..31
        const int nb = 32 * wc;
        V16 bg0, bg1, bu0, bu1;
        const bf16* p0 = &lBg[(nb +      l) * LROW + hi * 16];
        bg0.h[0] = *(const v8bf*)(p0); bg0.h[1] = *(const v8bf*)(p0 + 8);
        const bf16* p1 = &lBg[(nb + 16 + l) * LROW + hi * 16];
        bg1.h[0] = *(const v8bf*)(p1); bg1.h[1] = *(const v8bf*)(p1 + 8);
        const bf16* p2 = &lBu[(nb +      l) * LROW + hi * 16];
        bu0.h[0] = *(const v8bf*)(p2); bu0.h[1] = *(const v8bf*)(p2 + 8);
        const bf16* p3 = &lBu[(nb + 16 + l) * LROW + hi * 16];
        bu1.h[0] = *(const v8bf*)(p3); bu1.h[1] = *(const v8bf*)(p3 + 8);
        accg0 = __builtin_amdgcn_wmma_f32_16x16x32_bf16(false, af.v, false, bg0.v, (short)0, accg0, false, false);
        accg1 = __builtin_amdgcn_wmma_f32_16x16x32_bf16(false, af.v, false, bg1.v, (short)0, accg1, false, false);
        accu0 = __builtin_amdgcn_wmma_f32_16x16x32_bf16(false, af.v, false, bu0.v, (short)0, accu0, false, false);
        accu1 = __builtin_amdgcn_wmma_f32_16x16x32_bf16(false, af.v, false, bu1.v, (short)0, accu1, false, false);
    };

    loadRegs(0);
    for (int ks = 0; ks < KDIM / 32; ++ks) {
        __syncthreads();
        storeLds();
        __syncthreads();
        if (ks + 1 < KDIM / 32) loadRegs((ks + 1) * 32);   // overlap next slice with WMMA
        compute();
    }

    // epilogue: silu(gate)*up, scatter to h[slot, col]
#pragma unroll
    for (int c = 0; c < 2; ++c) {
        v8f ag = c ? accg1 : accg0;
        v8f au = c ? accu1 : accu0;
#pragma unroll
        for (int i = 0; i < 8; ++i) {
            int mr = 16 * wr + i + 8 * hi;
            int slot = s_slot[mr];
            if (slot >= 0) {
                float g = ag[i], u = au[i];
                float sig = 1.0f / (1.0f + __expf(-g));
                hbuf[(size_t)slot * IDIM + (n0 + 32 * wc + 16 * c + l)] = g * sig * u;
            }
        }
    }
}

// ---------------- GEMM2: partial[slot, 0:K] = h @ W2^T ---------------------
__global__ __launch_bounds__(128)
void moe_gemm2(const float* __restrict__ hbuf, const float* __restrict__ w2,
               const int* __restrict__ counts, const int* __restrict__ lists,
               float* __restrict__ partial) {
    const int e = blockIdx.z;
    const int cnt = counts[e];
    const int row0 = blockIdx.y * 32;
    if (row0 >= cnt) return;
    const int n0 = blockIdx.x * 64;   // output columns over K

    __shared__ __align__(16) bf16 lA[32 * LROW];
    __shared__ __align__(16) bf16 lB[64 * LROW];
    __shared__ int s_slot[32];
    __shared__ int s_row[32];

    const int tid = threadIdx.x;
    if (tid < 32) {
        int r = row0 + tid;
        int slot = (r < cnt) ? lists[e * SLOTS + r] : -1;
        s_slot[tid] = slot;
        s_row[tid]  = (slot >= 0) ? slot : 0;
    }
    __syncthreads();

    const int ar = tid >> 2;
    const int ac = (tid & 3) * 8;
    const int br = tid >> 1;
    const int bc = (tid & 1) * 16;
    const size_t aBase = (size_t)s_row[ar] * IDIM + ac;
    const size_t bBase = (size_t)e * KDIM * IDIM + (size_t)(n0 + br) * IDIM + bc;

    float4 ra0, ra1, rb0, rb1, rb2, rb3;
    auto loadRegs = [&](int i0) {
        const float* pa = hbuf + aBase + i0;
        ra0 = *(const float4*)(pa);     ra1 = *(const float4*)(pa + 4);
        const float* pb = w2 + bBase + i0;
        rb0 = *(const float4*)(pb);     rb1 = *(const float4*)(pb + 4);
        rb2 = *(const float4*)(pb + 8); rb3 = *(const float4*)(pb + 12);
    };
    auto storeLds = [&]() {
        *(v8bf*)&lA[ar * LROW + ac]     = cvt8(ra0, ra1);
        *(v8bf*)&lB[br * LROW + bc]     = cvt8(rb0, rb1);
        *(v8bf*)&lB[br * LROW + bc + 8] = cvt8(rb2, rb3);
    };

    const int lane = tid & 31;
    const int wid  = tid >> 5;
    const int wr = wid >> 1, wc = wid & 1;
    const int l  = lane & 15, hi = lane >> 4;

    v8f acc0 = {}, acc1 = {};
    auto compute = [&]() {
        V16 af;
        const bf16* pA = &lA[(16 * wr + l) * LROW];
        af.h[0] = *(const v8bf*)(pA + hi * 16);
        af.h[1] = *(const v8bf*)(pA + hi * 16 + 8);
        const int nb = 32 * wc;
        V16 b0, b1;
        const bf16* p0 = &lB[(nb +      l) * LROW + hi * 16];
        b0.h[0] = *(const v8bf*)(p0); b0.h[1] = *(const v8bf*)(p0 + 8);
        const bf16* p1 = &lB[(nb + 16 + l) * LROW + hi * 16];
        b1.h[0] = *(const v8bf*)(p1); b1.h[1] = *(const v8bf*)(p1 + 8);
        acc0 = __builtin_amdgcn_wmma_f32_16x16x32_bf16(false, af.v, false, b0.v, (short)0, acc0, false, false);
        acc1 = __builtin_amdgcn_wmma_f32_16x16x32_bf16(false, af.v, false, b1.v, (short)0, acc1, false, false);
    };

    loadRegs(0);
    for (int ks = 0; ks < IDIM / 32; ++ks) {
        __syncthreads();
        storeLds();
        __syncthreads();
        if (ks + 1 < IDIM / 32) loadRegs((ks + 1) * 32);
        compute();
    }

#pragma unroll
    for (int c = 0; c < 2; ++c) {
        v8f a = c ? acc1 : acc0;
#pragma unroll
        for (int i = 0; i < 8; ++i) {
            int mr = 16 * wr + i + 8 * hi;
            int slot = s_slot[mr];
            if (slot >= 0)
                partial[(size_t)slot * KDIM + (n0 + 32 * wc + 16 * c + l)] = a[i];
        }
    }
}

// ---------------- combine: out[m] = sum_k tw[m,k] * partial[slot(m,k)] ------
__global__ void moe_combine(const float* __restrict__ partial,
                            const float* __restrict__ tw, float* __restrict__ out) {
    size_t idx = (size_t)blockIdx.x * blockDim.x + threadIdx.x;   // over M*K/4
    if (idx >= (size_t)MDIM * KDIM / 4) return;
    int m  = (int)(idx / (KDIM / 4));
    int c4 = (int)(idx % (KDIM / 4)) * 4;
    const float4 p0 = *(const float4*)&partial[(size_t)(2 * m)     * KDIM + c4];
    const float4 p1 = *(const float4*)&partial[(size_t)(2 * m + 1) * KDIM + c4];
    float wA = tw[2 * m], wB = tw[2 * m + 1];
    float4 o;
    o.x = wA * p0.x + wB * p1.x;
    o.y = wA * p0.y + wB * p1.y;
    o.z = wA * p0.z + wB * p1.z;
    o.w = wA * p0.w + wB * p1.w;
    *(float4*)&out[(size_t)m * KDIM + c4] = o;
}

extern "C" void kernel_launch(void* const* d_in, const int* in_sizes, int n_in,
                              void* d_out, int out_size, void* d_ws, size_t ws_size,
                              hipStream_t stream) {
    (void)in_sizes; (void)n_in; (void)out_size; (void)ws_size;
    const float* x   = (const float*)d_in[0];
    const float* w1  = (const float*)d_in[1];
    const float* w2  = (const float*)d_in[2];
    const float* tw  = (const float*)d_in[3];
    const int*   ids = (const int*)d_in[4];
    float* out = (float*)d_out;

    char* ws = (char*)d_ws;
    int* counts = (int*)ws;                                   // 8 ints (256B slab)
    int* lists  = (int*)(ws + 256);                           // E*SLOTS ints
    float* hbuf = (float*)(ws + 256 + (size_t)ENUM * SLOTS * sizeof(int)); // SLOTS*I
    float* partial = hbuf + (size_t)SLOTS * IDIM;             // SLOTS*K

    hipMemsetAsync(counts, 0, 256, stream);
    moe_route<<<(SLOTS + 255) / 256, 256, 0, stream>>>(ids, counts, lists);

    dim3 g1(IDIM / 64, SLOTS / 32, ENUM);
    moe_gemm1<<<g1, 128, 0, stream>>>(x, w1, counts, lists, hbuf);

    dim3 g2(KDIM / 64, SLOTS / 32, ENUM);
    moe_gemm2<<<g2, 128, 0, stream>>>(hbuf, w2, counts, lists, partial);

    moe_combine<<<((size_t)MDIM * KDIM / 4 + 255) / 256, 256, 0, stream>>>(partial, tw, out);
}
